// Attention_90623809945662
// MI455X (gfx1250) — matrix operand order repaired and checked by reading, not compiled, not run on hardware
//
#include <hip/hip_runtime.h>
#include <hip/hip_bf16.h>

// ---------------------------------------------------------------------------
// Attention (B=4, N=2048, C=768, H=12, D=64) for gfx1250 (CDNA5, wave32)
// f16 data / f32 WMMA accum, flash-attention with async-LDS K/V double buffer
// ---------------------------------------------------------------------------

typedef __attribute__((ext_vector_type(16))) _Float16 v16h;
typedef __attribute__((ext_vector_type(8)))  _Float16 v8h;
typedef __attribute__((ext_vector_type(4)))  _Float16 v4h;
typedef __attribute__((ext_vector_type(8)))  float    v8f;

#define DIMC     768
#define HEADS    12
#define HEAD_DIM 64
#define BATCH    4
#define SEQ      2048
#define ROWS     (BATCH * SEQ)   // 8192
#define TK       64              // key tile (flash)

union H16 { v16h v; v8h h[2]; };

// --- WMMA fragment loaders (contiguous k order, same for A and B) -------
// A 16x32 f16: lane L row (L%16); halfs 0..15 = k0+(L<16?0:16)+e.
__device__ __forceinline__ v16h load_a_frag(const _Float16* base, int ld,
                                            int row0, int k0, int lane) {
  int r  = row0 + (lane & 15);
  int kk = k0 + ((lane & 16) ? 16 : 0);
  const _Float16* p = base + (size_t)r * ld + kk;
  H16 u;
  u.h[0] = *(const v8h*)(p);
  u.h[1] = *(const v8h*)(p + 8);
  return u.v;
}

// B 32x16 f16: lane L column (L%16); halfs 0..15 = k0+(L<16?0:16)+e.
// `base` laid out so one column is contiguous in K.
__device__ __forceinline__ v16h load_b_frag(const _Float16* base, int ld,
                                            int n0, int k0, int lane) {
  int n  = n0 + (lane & 15);
  int kk = k0 + ((lane & 16) ? 16 : 0);
  return *(const v16h*)(base + (size_t)n * ld + kk);
}

__device__ __forceinline__ v8f wmma_f16(v16h a, v16h b, v8f c) {
  return __builtin_amdgcn_wmma_f32_16x16x32_f16(false, a, false, b,
                                                (short)0, c, false, false);
}

// --- Async global->LDS staging of one 64x64 K tile + 64x64 V tile ----------
// 256 threads x 32B each per matrix (4 async b128 ops/thread total).
// INST_OFFSET applies to both LDS and global addresses (ISA §10 async ops).
__device__ __forceinline__ void stage_kv_async(
    const _Float16* __restrict__ Kg, const _Float16* __restrict__ Vg,
    unsigned kbase, unsigned vbase, int m0, int tid) {
  {
    // K tile: rows m0..m0+63, d 0..63 — contiguous 8KB in global.
    unsigned long long g = (unsigned long long)(uintptr_t)(Kg + (size_t)m0 * HEAD_DIM + tid * 16);
    unsigned l = kbase + (unsigned)tid * 32u;
    asm volatile("global_load_async_to_lds_b128 %0, %1, off"
                 :: "v"(l), "v"(g) : "memory");
    asm volatile("global_load_async_to_lds_b128 %0, %1, off offset:16"
                 :: "v"(l), "v"(g) : "memory");
  }
  {
    // V tile: Vt is [D, N]; 64 rows of 128B starting at column m0.
    int row = tid >> 2;
    int c   = (tid & 3) * 16;
    unsigned long long g = (unsigned long long)(uintptr_t)(Vg + (size_t)row * SEQ + m0 + c);
    unsigned l = vbase + (unsigned)(row * TK + c) * 2u;
    asm volatile("global_load_async_to_lds_b128 %0, %1, off"
                 :: "v"(l), "v"(g) : "memory");
    asm volatile("global_load_async_to_lds_b128 %0, %1, off offset:16"
                 :: "v"(l), "v"(g) : "memory");
  }
}

// --- Kernel 1: fp32 -> f16 convert -----------------------------------------
__global__ __launch_bounds__(256) void f32_to_f16_kernel(
    const float* __restrict__ in, _Float16* __restrict__ out, int n) {
  int i = (blockIdx.x * 256 + threadIdx.x) * 4;
  if (i + 3 < n) {
    float4 v = *(const float4*)(in + i);
    v4h h;
    h.x = (_Float16)v.x; h.y = (_Float16)v.y;
    h.z = (_Float16)v.z; h.w = (_Float16)v.w;
    *(v4h*)(out + i) = h;
  }
}

// --- Kernel 2: transpose+convert the four 768x768 weights ------------------
// T[n][k] = (f16) W[k][n]
__global__ __launch_bounds__(256) void transpose_w_kernel(
    const float* __restrict__ W0, const float* __restrict__ W1,
    const float* __restrict__ W2, const float* __restrict__ W3,
    _Float16* __restrict__ T0, _Float16* __restrict__ T1,
    _Float16* __restrict__ T2, _Float16* __restrict__ T3) {
  __shared__ float tile[32][33];
  const float* W; _Float16* T;
  switch (blockIdx.z) {
    case 0: W = W0; T = T0; break;
    case 1: W = W1; T = T1; break;
    case 2: W = W2; T = T2; break;
    default: W = W3; T = T3; break;
  }
  int bx = blockIdx.x * 32;   // n block
  int by = blockIdx.y * 32;   // k block
  int tx = threadIdx.x, ty = threadIdx.y;
#pragma unroll
  for (int i = 0; i < 32; i += 8)
    tile[ty + i][tx] = W[(size_t)(by + ty + i) * DIMC + (bx + tx)];
  __syncthreads();
#pragma unroll
  for (int i = 0; i < 32; i += 8)
    T[(size_t)(bx + ty + i) * DIMC + (by + tx)] = (_Float16)tile[tx][ty + i];
}

// --- Kernel 3: QKV projection GEMM -----------------------------------------
// 8 waves/block; wave tile 64x32 (8 WMMA per K=32 step); block 128x128.
// z=0: Q*0.125 -> [B,H,N,D]; z=1: K -> [B,H,N,D]; z=2: V -> [B,H,D,N]
__global__ __launch_bounds__(256) void qkv_gemm_kernel(
    const _Float16* __restrict__ X,
    const _Float16* __restrict__ Wqt, const _Float16* __restrict__ Wkt,
    const _Float16* __restrict__ Wvt,
    _Float16* __restrict__ Q, _Float16* __restrict__ Kd,
    _Float16* __restrict__ V) {
  int lane = threadIdx.x & 31;
  int wave = threadIdx.x >> 5;
  int m0 = blockIdx.y * 128 + (wave & 1) * 64;
  int n0 = blockIdx.x * 128 + (wave >> 1) * 32;
  int which = blockIdx.z;
  const _Float16* W = (which == 0) ? Wqt : (which == 1) ? Wkt : Wvt;

  v8f acc[4][2];
#pragma unroll
  for (int i = 0; i < 4; ++i)
#pragma unroll
    for (int j = 0; j < 2; ++j)
      acc[i][j] = (v8f){0.f,0.f,0.f,0.f,0.f,0.f,0.f,0.f};

  for (int k0 = 0; k0 < DIMC; k0 += 32) {
    v16h a[4], b[2];
#pragma unroll
    for (int i = 0; i < 4; ++i) a[i] = load_a_frag(X, DIMC, m0 + 16 * i, k0, lane);
#pragma unroll
    for (int j = 0; j < 2; ++j) b[j] = load_b_frag(W, DIMC, n0 + 16 * j, k0, lane);
#pragma unroll
    for (int i = 0; i < 4; ++i)
#pragma unroll
      for (int j = 0; j < 2; ++j)
        acc[i][j] = wmma_f16(a[i], b[j], acc[i][j]);
  }

  _Float16* Out = (which == 0) ? Q : (which == 1) ? Kd : V;
  float scale = (which == 0) ? 0.125f : 1.0f;  // SCALE = 64^-0.5, exact pow2
#pragma unroll
  for (int i = 0; i < 4; ++i)
#pragma unroll
    for (int j = 0; j < 2; ++j) {
      int col  = n0 + j * 16 + (lane & 15);
      int h    = col >> 6, d = col & 63;
      int rowb = m0 + i * 16 + ((lane & 16) ? 8 : 0);
#pragma unroll
      for (int r = 0; r < 8; ++r) {
        int m = rowb + r;
        int b_ = m >> 11, n = m & 2047;
        float val = acc[i][j][r] * scale;
        size_t idx;
        if (which < 2)
          idx = (((size_t)(b_ * HEADS + h)) * SEQ + n) * HEAD_DIM + d;
        else
          idx = (((size_t)(b_ * HEADS + h)) * HEAD_DIM + d) * SEQ + n;
        Out[idx] = (_Float16)val;
      }
    }
}

// --- Kernel 4: flash attention ---------------------------------------------
// grid (SEQ/128, B*H). 8 waves; 16 query rows/wave; 64-key tiles staged in
// LDS via async global->LDS loads, double-buffered.
__global__ __launch_bounds__(256) void flash_attn_kernel(
    const _Float16* __restrict__ Q, const _Float16* __restrict__ K,
    const _Float16* __restrict__ Vt, _Float16* __restrict__ O) {
  __shared__ _Float16 kv[2][2][TK * HEAD_DIM];  // [buf][K|V][4096] = 32KB
  __shared__ _Float16 psh[8][16 * TK];          // per-wave P tile   = 16KB

  int tid  = threadIdx.x;
  int lane = tid & 31;
  int wave = tid >> 5;
  int bh = blockIdx.y;
  int n0 = blockIdx.x * 128 + wave * 16;
  const _Float16* Qb = Q  + (size_t)bh * SEQ * HEAD_DIM;
  const _Float16* Kb = K  + (size_t)bh * SEQ * HEAD_DIM;
  const _Float16* Vb = Vt + (size_t)bh * HEAD_DIM * SEQ;

  // LDS byte offsets (low 32 bits of generic address = LDS offset)
  unsigned kbase[2], vbase[2];
  kbase[0] = (unsigned)(uintptr_t)&kv[0][0][0];
  vbase[0] = (unsigned)(uintptr_t)&kv[0][1][0];
  kbase[1] = (unsigned)(uintptr_t)&kv[1][0][0];
  vbase[1] = (unsigned)(uintptr_t)&kv[1][1][0];

  stage_kv_async(Kb, Vb, kbase[0], vbase[0], 0, tid);   // prefetch tile 0

  // Q fragments (rows n0..n0+15, d 0..63), pre-scaled by 1/8
  v16h qa0 = load_a_frag(Qb, HEAD_DIM, n0, 0,  lane);
  v16h qa1 = load_a_frag(Qb, HEAD_DIM, n0, 32, lane);

  v8f o[4];
#pragma unroll
  for (int j = 0; j < 4; ++j) o[j] = (v8f){0.f,0.f,0.f,0.f,0.f,0.f,0.f,0.f};
  float mrun[8], lrun[8];
#pragma unroll
  for (int r = 0; r < 8; ++r) { mrun[r] = -__builtin_inff(); lrun[r] = 0.f; }

  _Float16* pt = &psh[wave][0];
  int colw = lane & 15;
  int rowb = (lane & 16) ? 8 : 0;

  const int T = SEQ / TK;   // 32 tiles
  for (int t = 0; t < T; ++t) {
    int cur = t & 1;
    if (t + 1 < T) {
      stage_kv_async(Kb, Vb, kbase[cur ^ 1], vbase[cur ^ 1], (t + 1) * TK, tid);
      asm volatile("s_wait_asynccnt 0x4" ::: "memory");  // tile t done, t+1 pending
    } else {
      asm volatile("s_wait_asynccnt 0x0" ::: "memory");
    }
    __syncthreads();

    const _Float16* ksh = &kv[cur][0][0];
    const _Float16* vsh = &kv[cur][1][0];

    // S[16 x 64] = Q * K^T  (4 column tiles, K-dim = 64 over d)
    v8f s[4];
#pragma unroll
    for (int mj = 0; mj < 4; ++mj) s[mj] = (v8f){0.f,0.f,0.f,0.f,0.f,0.f,0.f,0.f};
#pragma unroll
    for (int mj = 0; mj < 4; ++mj) {
      v16h kb0 = load_b_frag(ksh, HEAD_DIM, mj * 16, 0,  lane);
      v16h kb1 = load_b_frag(ksh, HEAD_DIM, mj * 16, 32, lane);
      s[mj] = wmma_f16(qa0, kb0, s[mj]);
      s[mj] = wmma_f16(qa1, kb1, s[mj]);
    }

    // online softmax (rows 0..7 in lanes 0..15, rows 8..15 in lanes 16..31)
    float corr[8];
#pragma unroll
    for (int r = 0; r < 8; ++r) {
      float mx = fmaxf(fmaxf(s[0][r], s[1][r]), fmaxf(s[2][r], s[3][r]));
      mx = fmaxf(mx, __shfl_xor(mx, 1));
      mx = fmaxf(mx, __shfl_xor(mx, 2));
      mx = fmaxf(mx, __shfl_xor(mx, 4));
      mx = fmaxf(mx, __shfl_xor(mx, 8));
      float mnew = fmaxf(mrun[r], mx);
      corr[r] = __expf(mrun[r] - mnew);
      mrun[r] = mnew;
      float p0 = __expf(s[0][r] - mnew);
      float p1 = __expf(s[1][r] - mnew);
      float p2 = __expf(s[2][r] - mnew);
      float p3 = __expf(s[3][r] - mnew);
      s[0][r] = p0; s[1][r] = p1; s[2][r] = p2; s[3][r] = p3;
      float sum = (p0 + p1) + (p2 + p3);
      sum += __shfl_xor(sum, 1);
      sum += __shfl_xor(sum, 2);
      sum += __shfl_xor(sum, 4);
      sum += __shfl_xor(sum, 8);
      lrun[r] = lrun[r] * corr[r] + sum;
    }
#pragma unroll
    for (int j = 0; j < 4; ++j)
#pragma unroll
      for (int r = 0; r < 8; ++r) o[j][r] *= corr[r];

    // P: C-frag layout -> LDS [16][64] -> reread as two A-frags
#pragma unroll
    for (int r = 0; r < 8; ++r) {
      int ro = (rowb + r) * TK + colw;
      pt[ro +  0] = (_Float16)s[0][r];
      pt[ro + 16] = (_Float16)s[1][r];
      pt[ro + 32] = (_Float16)s[2][r];
      pt[ro + 48] = (_Float16)s[3][r];
    }
    asm volatile("" ::: "memory");   // keep DS store->load order (in-order per wave)
    v16h pa0 = load_a_frag(pt, TK, 0, 0,  lane);
    v16h pa1 = load_a_frag(pt, TK, 0, 32, lane);
    asm volatile("" ::: "memory");

    // O += P * V   (V tile in LDS is [d][m], columns contiguous in m)
#pragma unroll
    for (int j = 0; j < 4; ++j) {
      v16h vb0 = load_b_frag(vsh, TK, j * 16, 0,  lane);
      v16h vb1 = load_b_frag(vsh, TK, j * 16, 32, lane);
      o[j] = wmma_f16(pa0, vb0, o[j]);
      o[j] = wmma_f16(pa1, vb1, o[j]);
    }
    __syncthreads();   // all waves done with buf `cur` before it is re-staged
  }

  int b = bh / HEADS, h = bh % HEADS;
  int colbase = h * HEAD_DIM + (lane & 15);
  int rowg = b * SEQ + n0 + rowb;
#pragma unroll
  for (int r = 0; r < 8; ++r) {
    float inv = 1.0f / lrun[r];
    size_t rowoff = (size_t)(rowg + r) * DIMC + colbase;
    O[rowoff +  0] = (_Float16)(o[0][r] * inv);
    O[rowoff + 16] = (_Float16)(o[1][r] * inv);
    O[rowoff + 32] = (_Float16)(o[2][r] * inv);
    O[rowoff + 48] = (_Float16)(o[3][r] * inv);
  }
}

// --- Kernel 5: output projection + bias, fp32 out --------------------------
__global__ __launch_bounds__(256) void out_gemm_kernel(
    const _Float16* __restrict__ X, const _Float16* __restrict__ Wt,
    const float* __restrict__ bias, float* __restrict__ Out) {
  int lane = threadIdx.x & 31;
  int wave = threadIdx.x >> 5;
  int m0 = blockIdx.y * 128 + (wave & 1) * 64;
  int n0 = blockIdx.x * 128 + (wave >> 1) * 32;

  v8f acc[4][2];
#pragma unroll
  for (int i = 0; i < 4; ++i)
#pragma unroll
    for (int j = 0; j < 2; ++j)
      acc[i][j] = (v8f){0.f,0.f,0.f,0.f,0.f,0.f,0.f,0.f};

  for (int k0 = 0; k0 < DIMC; k0 += 32) {
    v16h a[4], b[2];
#pragma unroll
    for (int i = 0; i < 4; ++i) a[i] = load_a_frag(X, DIMC, m0 + 16 * i, k0, lane);
#pragma unroll
    for (int j = 0; j < 2; ++j) b[j] = load_b_frag(Wt, DIMC, n0 + 16 * j, k0, lane);
#pragma unroll
    for (int i = 0; i < 4; ++i)
#pragma unroll
      for (int j = 0; j < 2; ++j)
        acc[i][j] = wmma_f16(a[i], b[j], acc[i][j]);
  }

#pragma unroll
  for (int i = 0; i < 4; ++i)
#pragma unroll
    for (int j = 0; j < 2; ++j) {
      int col = n0 + j * 16 + (lane & 15);
      float bb = bias[col];
      int rowbase = m0 + i * 16 + ((lane & 16) ? 8 : 0);
#pragma unroll
      for (int r = 0; r < 8; ++r)
        Out[(size_t)(rowbase + r) * DIMC + col] = acc[i][j][r] + bb;
    }
}

// ---------------------------------------------------------------------------
extern "C" void kernel_launch(void* const* d_in, const int* in_sizes, int n_in,
                              void* d_out, int out_size, void* d_ws, size_t ws_size,
                              hipStream_t stream) {
  const float* x  = (const float*)d_in[0];
  const float* Wq = (const float*)d_in[1];
  const float* Wk = (const float*)d_in[2];
  const float* Wv = (const float*)d_in[3];
  const float* Wp = (const float*)d_in[4];
  const float* bp = (const float*)d_in[5];

  const size_t XE = (size_t)ROWS * DIMC;   // 6,291,456 elems
  const size_t WE = (size_t)DIMC * DIMC;   //   589,824 elems

  _Float16* ws  = (_Float16*)d_ws;
  _Float16* x16 = ws;
  _Float16* wqt = x16 + XE;
  _Float16* wkt = wqt + WE;
  _Float16* wvt = wkt + WE;
  _Float16* wpt = wvt + WE;
  _Float16* q16 = wpt + WE;
  _Float16* k16 = q16 + XE;
  _Float16* v16 = k16 + XE;
  _Float16* o16 = v16 + XE;

  f32_to_f16_kernel<<<dim3((unsigned)(XE / 4 / 256)), dim3(256), 0, stream>>>(
      x, x16, (int)XE);
  transpose_w_kernel<<<dim3(24, 24, 4), dim3(32, 8), 0, stream>>>(
      Wq, Wk, Wv, Wp, wqt, wkt, wvt, wpt);
  qkv_gemm_kernel<<<dim3(6, 64, 3), dim3(256), 0, stream>>>(
      x16, wqt, wkt, wvt, q16, k16, v16);
  flash_attn_kernel<<<dim3(SEQ / 128, BATCH * HEADS), dim3(256), 0, stream>>>(
      q16, k16, v16, o16);
  out_gemm_kernel<<<dim3(6, 64), dim3(256), 0, stream>>>(
      o16, wpt, bp, (float*)d_out);
}
